// GNN_39298950758409
// MI455X (gfx1250) — matrix operand order repaired and checked by reading, not compiled, not run on hardware
//
#include <hip/hip_runtime.h>
#include <hip/hip_bf16.h>

#define D128 128

typedef __attribute__((ext_vector_type(16))) __bf16        v16bf;
typedef __attribute__((ext_vector_type(8)))  float         v8f;
typedef __attribute__((ext_vector_type(4)))  unsigned int  v4u;

union BF16Frag {
    v16bf v;
    v4u   q[2];
};

// float -> bf16 (round to nearest even), stored as raw u16
__device__ __forceinline__ unsigned short f2bf(float f) {
    unsigned int u = __float_as_uint(f);
    u += 0x7FFFu + ((u >> 16) & 1u);
    return (unsigned short)(u >> 16);
}

__device__ __forceinline__ float silu(float x) {
    return x / (1.0f + __expf(-x));
}

// A fragment: 16x32 bf16 strip for one lane; rowp points to row (m0+lane&15),
// K-contiguous bf16. half = lane>>4.
__device__ __forceinline__ v16bf load_a(const unsigned short* rowp, int k0, int half) {
    BF16Frag f;
    f.q[0] = *(const v4u*)(rowp + k0 + half * 8);
    f.q[1] = *(const v4u*)(rowp + k0 + 16 + half * 8);
    return f.v;
}

// B fragment: weights pre-transposed to [out][in] bf16; lane n holds K ranges
// k0 + half*8 .. +7 and k0 + 16 + half*8 .. +7 of column (ct*16 + n).
__device__ __forceinline__ v16bf load_b(const unsigned short* wt, int Kdim, int col,
                                        int k0, int half) {
    const unsigned short* p = wt + (size_t)col * Kdim + k0 + half * 8;
    BF16Frag f;
    f.q[0] = *(const v4u*)(p);
    f.q[1] = *(const v4u*)(p + 16);
    return f.v;
}

// ---------------------------------------------------------------- prep kernels
__global__ void GNN_zero_kernel(float* p, long n) {
    long i = (long)blockIdx.x * blockDim.x + threadIdx.x;
    long stride = (long)gridDim.x * blockDim.x;
    for (; i < n; i += stride) p[i] = 0.0f;
}

// w: [K][Nc] f32 row-major  ->  wt: [Nc][K] bf16 row-major
__global__ void GNN_wconv_kernel(const float* __restrict__ w,
                                 unsigned short* __restrict__ wt, int K, int Nc) {
    int i = blockIdx.x * blockDim.x + threadIdx.x;
    if (i < K * Nc) {
        int k = i / Nc, n = i - k * Nc;
        wt[(size_t)n * K + k] = f2bf(w[(size_t)k * Nc + n]);
    }
}

// ---------------------------------------------------------------- edge kernel
// 64 edges/block, 128 threads = 4 wave32, each wave owns a 16-row strip.
__global__ __launch_bounds__(128, 1) void GNN_edge_kernel(
    const float* __restrict__ V, const float* __restrict__ E,
    const int* __restrict__ edges,
    const unsigned short* __restrict__ w1t, const float* __restrict__ b1,
    const unsigned short* __restrict__ w2t, const float* __restrict__ b2,
    float* __restrict__ edge_out,
    float* __restrict__ sum0, float* __restrict__ sum1,
    float* __restrict__ cnt0, float* __restrict__ cnt1, int M) {
    __shared__ __align__(16) unsigned short Xs[64][3 * D128];  // 48 KB
    __shared__ __align__(16) unsigned short Hs[64][D128];      // 16 KB

    const int tid  = threadIdx.x;
    const int base = blockIdx.x * 64;

    // per-edge counts for scatter-mean denominators
    if (tid < 64) {
        int g = base + tid;
        if (g < M) {
            atomicAdd(&cnt0[edges[2 * g + 0]], 1.0f);
            atomicAdd(&cnt1[edges[2 * g + 1]], 1.0f);
        }
    }

    // build X = [V[s] | V[r] | E] as bf16, coalesced across feature dim
    for (int row = 0; row < 64; ++row) {
        int g = base + row;
        if (g < M) {
            int s = edges[2 * g + 0];
            int r = edges[2 * g + 1];
            Xs[row][tid]            = f2bf(V[(size_t)s * D128 + tid]);
            Xs[row][D128 + tid]     = f2bf(V[(size_t)r * D128 + tid]);
            Xs[row][2 * D128 + tid] = f2bf(E[(size_t)g * D128 + tid]);
        } else {
            Xs[row][tid] = 0; Xs[row][D128 + tid] = 0; Xs[row][2 * D128 + tid] = 0;
        }
    }
    __syncthreads();

    const int lane = tid & 31, wid = tid >> 5;
    const int half = lane >> 4, nco = lane & 15;
    const int m0 = wid * 16;

    v8f zero = {0.f, 0.f, 0.f, 0.f, 0.f, 0.f, 0.f, 0.f};
    v8f acc[8];
#pragma unroll
    for (int ct = 0; ct < 8; ++ct) acc[ct] = zero;

    // GEMM1: [16 x 384] x [384 x 128]
    for (int k0 = 0; k0 < 3 * D128; k0 += 32) {
        v16bf a = load_a(Xs[m0 + nco], k0, half);
#pragma unroll
        for (int ct = 0; ct < 8; ++ct) {
            v16bf b = load_b(w1t, 3 * D128, ct * 16 + nco, k0, half);
            acc[ct] = __builtin_amdgcn_wmma_f32_16x16x32_bf16(
                false, a, false, b, (short)0, acc[ct], false, false);
        }
    }

    // bias + SiLU -> bf16 H strip (same wave produces & consumes its strip)
#pragma unroll
    for (int ct = 0; ct < 8; ++ct) {
        float bias = b1[ct * 16 + nco];
#pragma unroll
        for (int r = 0; r < 8; ++r) {
            float x = acc[ct][r] + bias;
            Hs[m0 + r + 8 * half][ct * 16 + nco] = f2bf(silu(x));
        }
    }

    v8f acc2[8];
#pragma unroll
    for (int ct = 0; ct < 8; ++ct) acc2[ct] = zero;

    // GEMM2: [16 x 128] x [128 x 128]
    for (int k0 = 0; k0 < D128; k0 += 32) {
        v16bf a = load_a(Hs[m0 + nco], k0, half);
#pragma unroll
        for (int ct = 0; ct < 8; ++ct) {
            v16bf b = load_b(w2t, D128, ct * 16 + nco, k0, half);
            acc2[ct] = __builtin_amdgcn_wmma_f32_16x16x32_bf16(
                false, a, false, b, (short)0, acc2[ct], false, false);
        }
    }

    float bb[8];
#pragma unroll
    for (int ct = 0; ct < 8; ++ct) bb[ct] = b2[ct * 16 + nco];

    // write edge embeddings + scatter-add halves into sum0/sum1
#pragma unroll
    for (int r = 0; r < 8; ++r) {
        int row = m0 + r + 8 * half;
        int g = base + row;
        if (g < M) {
            int s  = edges[2 * g + 0];
            int rr = edges[2 * g + 1];
#pragma unroll
            for (int ct = 0; ct < 8; ++ct) {
                int col = ct * 16 + nco;
                float val = acc2[ct][r] + bb[ct];
                edge_out[(size_t)g * D128 + col] = val;
                if (col < 64) atomicAdd(&sum0[(size_t)s * 64 + col], val);
                else          atomicAdd(&sum1[(size_t)rr * 64 + (col - 64)], val);
            }
        }
    }
}

// ---------------------------------------------------------------- node kernel
__global__ __launch_bounds__(128, 1) void GNN_node_kernel(
    const float* __restrict__ V,
    const unsigned short* __restrict__ w1t, const float* __restrict__ b1,
    const unsigned short* __restrict__ w2t, const float* __restrict__ b2,
    const float* __restrict__ sum0, const float* __restrict__ sum1,
    const float* __restrict__ cnt0, const float* __restrict__ cnt1,
    float* __restrict__ node_out, int N) {
    __shared__ __align__(16) unsigned short Xs[64][2 * D128];  // 32 KB
    __shared__ __align__(16) unsigned short Hs[64][D128];      // 16 KB

    const int tid  = threadIdx.x;
    const int base = blockIdx.x * 64;

    // build X = [V | sum0/cnt0 | sum1/cnt1] as bf16
    for (int row = 0; row < 64; ++row) {
        int g = base + row;
        if (g < N) {
            Xs[row][tid] = f2bf(V[(size_t)g * D128 + tid]);
            if (tid < 64) {
                float c = fmaxf(cnt0[g], 1.0f);
                Xs[row][D128 + tid] = f2bf(sum0[(size_t)g * 64 + tid] / c);
            } else {
                int t = tid - 64;
                float c = fmaxf(cnt1[g], 1.0f);
                Xs[row][D128 + 64 + t] = f2bf(sum1[(size_t)g * 64 + t] / c);
            }
        } else {
            Xs[row][tid] = 0; Xs[row][D128 + tid] = 0;
        }
    }
    __syncthreads();

    const int lane = tid & 31, wid = tid >> 5;
    const int half = lane >> 4, nco = lane & 15;
    const int m0 = wid * 16;

    v8f zero = {0.f, 0.f, 0.f, 0.f, 0.f, 0.f, 0.f, 0.f};
    v8f acc[8];
#pragma unroll
    for (int ct = 0; ct < 8; ++ct) acc[ct] = zero;

    // GEMM1: [16 x 256] x [256 x 128]
    for (int k0 = 0; k0 < 2 * D128; k0 += 32) {
        v16bf a = load_a(Xs[m0 + nco], k0, half);
#pragma unroll
        for (int ct = 0; ct < 8; ++ct) {
            v16bf b = load_b(w1t, 2 * D128, ct * 16 + nco, k0, half);
            acc[ct] = __builtin_amdgcn_wmma_f32_16x16x32_bf16(
                false, a, false, b, (short)0, acc[ct], false, false);
        }
    }

#pragma unroll
    for (int ct = 0; ct < 8; ++ct) {
        float bias = b1[ct * 16 + nco];
#pragma unroll
        for (int r = 0; r < 8; ++r) {
            float x = acc[ct][r] + bias;
            Hs[m0 + r + 8 * half][ct * 16 + nco] = f2bf(silu(x));
        }
    }

    v8f acc2[8];
#pragma unroll
    for (int ct = 0; ct < 8; ++ct) acc2[ct] = zero;

    // GEMM2: [16 x 128] x [128 x 128]
    for (int k0 = 0; k0 < D128; k0 += 32) {
        v16bf a = load_a(Hs[m0 + nco], k0, half);
#pragma unroll
        for (int ct = 0; ct < 8; ++ct) {
            v16bf b = load_b(w2t, D128, ct * 16 + nco, k0, half);
            acc2[ct] = __builtin_amdgcn_wmma_f32_16x16x32_bf16(
                false, a, false, b, (short)0, acc2[ct], false, false);
        }
    }

#pragma unroll
    for (int r = 0; r < 8; ++r) {
        int row = m0 + r + 8 * half;
        int g = base + row;
        if (g < N) {
#pragma unroll
            for (int ct = 0; ct < 8; ++ct) {
                int col = ct * 16 + nco;
                node_out[(size_t)g * D128 + col] = acc2[ct][r] + b2[col];
            }
        }
    }
}

// ---------------------------------------------------------------- launcher
extern "C" void kernel_launch(void* const* d_in, const int* in_sizes, int n_in,
                              void* d_out, int out_size, void* d_ws, size_t ws_size,
                              hipStream_t stream) {
    const float* V     = (const float*)d_in[0];
    const float* E     = (const float*)d_in[1];
    const int*   edges = (const int*)d_in[2];
    const float* fe_w1 = (const float*)d_in[3];
    const float* fe_b1 = (const float*)d_in[4];
    const float* fe_w2 = (const float*)d_in[5];
    const float* fe_b2 = (const float*)d_in[6];
    const float* fn_w1 = (const float*)d_in[7];
    const float* fn_b1 = (const float*)d_in[8];
    const float* fn_w2 = (const float*)d_in[9];
    const float* fn_b2 = (const float*)d_in[10];

    const int N = in_sizes[0] / D128;  // nodes
    const int M = in_sizes[2] / 2;     // edges

    float* node_out = (float*)d_out;
    float* edge_out = node_out + (size_t)N * D128;

    // workspace layout
    char* ws = (char*)d_ws;
    unsigned short* w1e_t = (unsigned short*)ws; ws += (size_t)D128 * 3 * D128 * 2;  // 98304
    unsigned short* w2e_t = (unsigned short*)ws; ws += (size_t)D128 * D128 * 2;      // 32768
    unsigned short* w1n_t = (unsigned short*)ws; ws += (size_t)D128 * 2 * D128 * 2;  // 65536
    unsigned short* w2n_t = (unsigned short*)ws; ws += (size_t)D128 * D128 * 2;      // 32768
    float* sum0 = (float*)ws; ws += (size_t)N * 64 * sizeof(float);
    float* sum1 = (float*)ws; ws += (size_t)N * 64 * sizeof(float);
    float* cnt0 = (float*)ws; ws += (size_t)N * sizeof(float);
    float* cnt1 = (float*)ws; ws += (size_t)N * sizeof(float);

    // 1) zero accumulators (sum0|sum1|cnt0|cnt1 contiguous: N*130 floats)
    {
        long n = (long)N * 130;
        GNN_zero_kernel<<<2048, 256, 0, stream>>>(sum0, n);
    }

    // 2) weight transpose+convert to bf16 [out][in]
    GNN_wconv_kernel<<<(3 * D128 * D128 + 255) / 256, 256, 0, stream>>>(fe_w1, w1e_t, 3 * D128, D128);
    GNN_wconv_kernel<<<(D128 * D128 + 255) / 256, 256, 0, stream>>>(fe_w2, w2e_t, D128, D128);
    GNN_wconv_kernel<<<(2 * D128 * D128 + 255) / 256, 256, 0, stream>>>(fn_w1, w1n_t, 2 * D128, D128);
    GNN_wconv_kernel<<<(D128 * D128 + 255) / 256, 256, 0, stream>>>(fn_w2, w2n_t, D128, D128);

    // 3) fused gather + edge MLP + scatter-add
    GNN_edge_kernel<<<(M + 63) / 64, 128, 0, stream>>>(
        V, E, edges, w1e_t, fe_b1, w2e_t, fe_b2,
        edge_out, sum0, sum1, cnt0, cnt1, M);

    // 4) fused mean + node MLP
    GNN_node_kernel<<<(N + 63) / 64, 128, 0, stream>>>(
        V, w1n_t, fn_b1, w2n_t, fn_b2, sum0, sum1, cnt0, cnt1, node_out, N);
}
